// LatentDecoder_76192719831699
// MI455X (gfx1250) — matrix operand order repaired and checked by reading, not compiled, not run on hardware
//
#include <hip/hip_runtime.h>
#include <math.h>

// ---------------------------------------------------------------------------
// LatentDecoder for MI455X (gfx1250, wave32). Per-edge / per-node GEMMs are
// mapped onto v_wmma_f32_16x16x32_f16 (16 rows per wave, f32 accumulate);
// weights are pre-swizzled in LDS to WMMA fragment order so B fragments are
// single 32-byte ds reads; A fragments are gathered as float4 + cvt_pk.
// Geometry, softmax and SO(3) rotations run on VALU.
// ---------------------------------------------------------------------------

typedef __attribute__((ext_vector_type(16))) _Float16 v16h;
typedef __attribute__((ext_vector_type(8)))  float    v8f;
typedef __attribute__((ext_vector_type(4)))  float    f4;

#define DEVFN __device__ __forceinline__

constexpr int   NN_  = 8192;
constexpr int   KNB  = 30;          // neighbors per node
constexpr int   EE   = NN_ * KNB;   // edges = 245760
constexpr int   NH_  = 8;
constexpr int   AC_  = 91;
constexpr int   NAA_ = 20;
constexpr int   LL_  = 4;
constexpr float EPS_ = 1e-7f;

// ---------------------------------------------------------------------------
// small helpers
// ---------------------------------------------------------------------------
DEVFN float ntn(float x) {            // nan_to_num
  if (x != x) return 0.f;
  if (x >  3.4e38f) return  3.4e38f;
  if (x < -3.4e38f) return -3.4e38f;
  return x;
}
DEVFN void cross3(const float* a, const float* b, float* o) {
  o[0] = a[1]*b[2] - a[2]*b[1];
  o[1] = a[2]*b[0] - a[0]*b[2];
  o[2] = a[0]*b[1] - a[1]*b[0];
}
DEVFN float dot3(const float* a, const float* b) {
  return a[0]*b[0] + a[1]*b[1] + a[2]*b[2];
}
DEVFN void norm3(const float* a, float* o) {   // v / sqrt(|v|^2 + EPS)
  float inv = 1.f / sqrtf(dot3(a, a) + EPS_);
  o[0] = a[0]*inv; o[1] = a[1]*inv; o[2] = a[2]*inv;
}

// ---------------------------------------------------------------------------
// WMMA fragment builders (CDNA5 16-bit 16x16x32 layouts, wave32)
// A 16x32: lane m=lane&15; lanes>=16 use k-base +8; vgpr v in 0..3 -> K=kb+2v,
// vgpr 4..7 -> K=16+kb+2v.  `base` points at this row's element k0 (contiguous
// f32, 16B aligned).  Loads: 4x float4 -> 8x v_cvt_pk_f16_f32.
// ---------------------------------------------------------------------------
DEVFN v16h make_a_frag(const float* base, int lane) {
  int kb = (lane >> 4) << 3;      // 0 or 8
  const f4* p0 = (const f4*)(base + kb);
  const f4* p1 = (const f4*)(base + 16 + kb);
  f4 x0 = p0[0], x1 = p0[1];
  f4 x2 = p1[0], x3 = p1[1];
  v16h a;
#pragma unroll
  for (int i = 0; i < 4; ++i) {
    a[i]      = (_Float16)x0[i];
    a[4 + i]  = (_Float16)x1[i];
    a[8 + i]  = (_Float16)x2[i];
    a[12 + i] = (_Float16)x3[i];
  }
  return a;
}

// B fragments are stored pre-swizzled in LDS: for (kt, nt), lane l holds 16
// halves covering K = kt*32 + ((i&8)<<1) + ((l>>4)<<3) + (i&7), N = (l&15) + nt*16.
DEVFN v16h load_b_swz(const _Float16* sW, int NT, int kt, int nt, int lane) {
  return *(const v16h*)(sW + (size_t)(((kt * NT + nt) * 32 + lane) << 4));
}

// stage weights (Kin x Nout row-major f32) into swizzled LDS fragment order
template<int KT, int NT>
DEVFN void stage_w_swz(_Float16* sW, const float* Wg, int Nout, int tid) {
  constexpr int TOT = KT * NT * 512;     // 32 lanes * 16 elems per tile
  for (int i = tid; i < TOT; i += 256) {
    int idx  = i & 15;
    int l    = (i >> 4) & 31;
    int tile = i >> 9;
    int nt = tile % NT, kt = tile / NT;
    int n = (l & 15) + nt * 16;
    int k = kt * 32 + ((idx & 8) << 1) + ((l >> 4) << 3) + (idx & 7);
    sW[i] = (n < Nout) ? (_Float16)Wg[k * Nout + n] : (_Float16)0.f;
  }
}

DEVFN v8f wmma_acc(v16h a, v16h b, v8f c) {
  return __builtin_amdgcn_wmma_f32_16x16x32_f16(false, a, false, b,
                                                (short)0, c, false, false);
}

// ---------------------------------------------------------------------------
// KNN: brute force with LDS tiles, per-thread sorted top-30 (ascending d2)
// ---------------------------------------------------------------------------
__global__ __launch_bounds__(256)
void knn_kernel(const float* __restrict__ bb, int* __restrict__ src) {
  __shared__ float sx[256], sy[256], sz[256];
  int q = blockIdx.x * 256 + threadIdx.x;
  float qx = bb[(size_t)q*12 + 3], qy = bb[(size_t)q*12 + 4], qz = bb[(size_t)q*12 + 5];
  float bd[KNB]; int bi[KNB];
#pragma unroll
  for (int i = 0; i < KNB; ++i) { bd[i] = 3.4e38f; bi[i] = 0; }
  for (int tile = 0; tile < NN_; tile += 256) {
    int j = tile + threadIdx.x;
    __syncthreads();
    sx[threadIdx.x] = bb[(size_t)j*12 + 3];
    sy[threadIdx.x] = bb[(size_t)j*12 + 4];
    sz[threadIdx.x] = bb[(size_t)j*12 + 5];
    __syncthreads();
    for (int jj = 0; jj < 256; ++jj) {
      int idx = tile + jj;
      if (idx == q) continue;
      float dx = qx - sx[jj], dy = qy - sy[jj], dz = qz - sz[jj];
      float d2 = dx*dx + dy*dy + dz*dz;
      if (d2 < bd[KNB-1]) {
        int p = KNB - 1;
        while (p > 0 && bd[p-1] > d2) { bd[p] = bd[p-1]; bi[p] = bi[p-1]; --p; }
        bd[p] = d2; bi[p] = idx;
      }
    }
  }
  for (int k = 0; k < KNB; ++k) src[(size_t)q*KNB + k] = bi[k];
}

// ---------------------------------------------------------------------------
// edge geometry: rotation frame R (E x 9) and edge features ef (E x 32)
// ---------------------------------------------------------------------------
__global__ __launch_bounds__(256)
void geom_kernel(const float* __restrict__ bb, const int* __restrict__ src,
                 float* __restrict__ ef, float* __restrict__ Rm) {
  int e = blockIdx.x * 256 + threadIdx.x;
  if (e >= EE) return;
  int n = e / KNB;
  int s = src[e];
  float v[3], z[3], xr[3], x[3], y[3];
  for (int d = 0; d < 3; ++d)
    v[d] = bb[(size_t)s*12 + 3 + d] - bb[(size_t)n*12 + 3 + d];
  float dist = sqrtf(dot3(v, v) + EPS_);
  norm3(v, z);
  float ref[3] = {1.f, 0.f, 0.f};
  if (fabsf(z[0]) >= 0.9f) { ref[0] = 0.f; ref[1] = 1.f; }
  float pr = dot3(ref, z);
  for (int d = 0; d < 3; ++d) xr[d] = ref[d] - pr * z[d];
  norm3(xr, x);
  cross3(z, x, y);
  for (int d = 0; d < 3; ++d) {
    Rm[(size_t)e*9 + 0 + d] = x[d];
    Rm[(size_t)e*9 + 3 + d] = y[d];
    Rm[(size_t)e*9 + 6 + d] = z[d];
  }
  // RBF: mu = linspace(0,20,16), sigma = 20/16
  for (int i = 0; i < 16; ++i) {
    float t = (dist - (20.f/15.f) * (float)i) / 1.25f;
    ef[(size_t)e*32 + i] = expf(-t*t);
  }
  // positional embedding
  float dd = (float)(s - n);
  for (int j = 0; j < 8; ++j) {
    float fr = expf(-(float)(2*j) * (9.210340371976184f / 16.f));
    ef[(size_t)e*32 + 16 + j] = cosf(dd * fr);
    ef[(size_t)e*32 + 24 + j] = sinf(dd * fr);
  }
}

// ---------------------------------------------------------------------------
// backbone embedding: dihedrals + orientations + virtual CB -> (N,4,26)
// ---------------------------------------------------------------------------
__global__ __launch_bounds__(256)
void bbemb_kernel(const float* __restrict__ bb, float* __restrict__ emb) {
  int n = blockIdx.x * 256 + threadIdx.x;
  if (n >= NN_) return;
  float dih[3];
  for (int r = 0; r < 3; ++r) {
    int t = 3*n + r;
    float D = 0.f;
    if (t >= 1 && t <= 3*NN_ - 3) {
      int i = t - 1;
      float X[4][3];
      for (int a = 0; a < 4; ++a) {
        int g = i + a;
        for (int d = 0; d < 3; ++d)
          X[a][d] = bb[(size_t)(g/3)*12 + (g%3)*3 + d];
      }
      float d0[3], d1[3], d2[3], u2[3], u1[3], u0[3];
      for (int d = 0; d < 3; ++d) { d0[d]=X[1][d]-X[0][d]; d1[d]=X[2][d]-X[1][d]; d2[d]=X[3][d]-X[2][d]; }
      norm3(d0, u2); norm3(d1, u1); norm3(d2, u0);
      float c0[3], c1[3], n2[3], n1v[3];
      cross3(u2, u1, c0); cross3(u1, u0, c1);
      norm3(c0, n2); norm3(c1, n1v);
      float cosD = dot3(n2, n1v);
      cosD = fminf(1.f - EPS_, fmaxf(-1.f + EPS_, cosD));
      float sg = dot3(u2, n1v);
      float sign = (sg > 0.f) ? 1.f : ((sg < 0.f) ? -1.f : 0.f);
      D = sign * acosf(cosD);
    }
    dih[r] = D;
  }
  float* row0 = emb + (size_t)n*4*26;
  for (int c = 0; c < 26; ++c) row0[c] = 0.f;
  for (int r = 0; r < 3; ++r) { row0[r] = cosf(dih[r]); row0[3+r] = sinf(dih[r]); }
  // 7 vectors: bb-ca (4), forward, backward, virtual cb
  float ca[3], vec[7][3];
  for (int d = 0; d < 3; ++d) ca[d] = bb[(size_t)n*12 + 3 + d];
  for (int a = 0; a < 4; ++a)
    for (int d = 0; d < 3; ++d) vec[a][d] = bb[(size_t)n*12 + a*3 + d] - ca[d];
  float tmp[3];
  if (n < NN_ - 1) {
    for (int d = 0; d < 3; ++d) tmp[d] = bb[(size_t)(n+1)*12 + 3 + d] - ca[d];
    norm3(tmp, vec[4]);
  } else { vec[4][0]=vec[4][1]=vec[4][2]=0.f; }
  if (n > 0) {
    for (int d = 0; d < 3; ++d) tmp[d] = bb[(size_t)(n-1)*12 + 3 + d] - ca[d];
    norm3(tmp, vec[5]);
  } else { vec[5][0]=vec[5][1]=vec[5][2]=0.f; }
  float bv[3], cv[3], av[3];
  for (int d = 0; d < 3; ++d) {
    bv[d] = bb[(size_t)n*12 + 3 + d] - bb[(size_t)n*12 + 0 + d];
    cv[d] = bb[(size_t)n*12 + 6 + d] - bb[(size_t)n*12 + 3 + d];
  }
  cross3(bv, cv, av);
  for (int d = 0; d < 3; ++d)
    vec[6][d] = -0.58273431f*av[d] + 0.56802827f*bv[d] - 0.54067466f*cv[d];
  for (int j = 0; j < 3; ++j) {
    float* row = emb + (size_t)(n*4 + 1 + j)*26;
    for (int c = 0; c < 26; ++c) row[c] = (c < 7) ? ntn(vec[c][j]) : 0.f;
  }
}

// ---------------------------------------------------------------------------
// SO(3) conv (shared by embed head, Cin=26/Cout=32, and output head 32/91)
// one block (128 thr) per node, scalar math, LDS staging
// ---------------------------------------------------------------------------
__global__ __launch_bounds__(128)
void so3conv_kernel(const float* __restrict__ xin, const float* __restrict__ ef,
                    const float* __restrict__ Rm, const int* __restrict__ src,
                    const float* __restrict__ Wrad, const float* __restrict__ brad,
                    const float* __restrict__ Wgate, const float* __restrict__ W0,
                    const float* __restrict__ W1, float* __restrict__ out,
                    int Cin, int Cout) {
  __shared__ float sEf[32], sR[9], sXs[4*32], sRad[64], sS1[3*32];
  int t = threadIdx.x, n = blockIdx.x;
  float m0s = 0.f, m1s0 = 0.f, m1s1 = 0.f, m1s2 = 0.f;
  for (int k = 0; k < KNB; ++k) {
    int e = n*KNB + k;
    int s = src[e];
    __syncthreads();
    if (t < 32) sEf[t] = ef[(size_t)e*32 + t];
    if (t < 9)  sR[t]  = Rm[(size_t)e*9 + t];
    for (int i = t; i < 4*Cin; i += 128) sXs[i] = xin[(size_t)s*4*Cin + i];
    __syncthreads();
    if (t < 64) {
      float r = brad[t];
      for (int c = 0; c < 32; ++c) r += sEf[c] * Wrad[c*64 + t];
      sRad[t] = r / (1.f + expf(-r));   // silu
    }
    if (t < 3*Cin) {
      int i = t / Cin, c = t % Cin;
      sS1[i*Cin + c] = sR[i*3+0]*sXs[1*Cin+c] + sR[i*3+1]*sXs[2*Cin+c] + sR[i*3+2]*sXs[3*Cin+c];
    }
    __syncthreads();
    if (t < Cout) {
      float g0 = 0.f, g1 = 0.f;
      for (int c = 0; c < 64; ++c) {
        float rr = sRad[c];
        g0 += rr * Wgate[c*2*Cout + t];
        g1 += rr * Wgate[c*2*Cout + Cout + t];
      }
      float m0 = 0.f, m10 = 0.f, m11 = 0.f, m12 = 0.f;
      for (int c = 0; c < Cin; ++c) {
        m0  += sXs[c] * W0[c*Cout + t];
        float w = W1[c*Cout + t];
        m10 += sS1[0*Cin + c]*w; m11 += sS1[1*Cin + c]*w; m12 += sS1[2*Cin + c]*w;
      }
      m10 *= g1; m11 *= g1; m12 *= g1;
      m0s  += m0 * g0;
      m1s0 += sR[0]*m10 + sR[3]*m11 + sR[6]*m12;   // R^T rotate back
      m1s1 += sR[1]*m10 + sR[4]*m11 + sR[7]*m12;
      m1s2 += sR[2]*m10 + sR[5]*m11 + sR[8]*m12;
    }
  }
  if (t < Cout) {
    float sc = rsqrtf((float)KNB);
    out[((size_t)n*4 + 0)*Cout + t] = m0s  * sc;
    out[((size_t)n*4 + 1)*Cout + t] = m1s0 * sc;
    out[((size_t)n*4 + 2)*Cout + t] = m1s1 * sc;
    out[((size_t)n*4 + 3)*Cout + t] = m1s2 * sc;
  }
}

__global__ __launch_bounds__(256)
void initres_kernel(const float* __restrict__ bbout, const float* __restrict__ latent,
                    float* __restrict__ res) {
  int idx = blockIdx.x * 256 + threadIdx.x;   // N*4*64
  int c = idx & 63, row = idx >> 6;
  res[idx] = (c < 32) ? bbout[(size_t)row*32 + c] : latent[(size_t)row*32 + (c - 32)];
}

// ---------------------------------------------------------------------------
// WMMA GEMM over gathered 160-dim edge features:
//   feat(e) = [ res0[src[e]] (64) | res0[e/K] (64) | ef[e] (32) ]
// NT = output 16-col tiles; EPI: 0 = store, 2 = tanh(x + bias)
// ---------------------------------------------------------------------------
template<int NT, int EPI>
__global__ __launch_bounds__(256)
void gemm_feat_kernel(const float* __restrict__ res, const float* __restrict__ ef,
                      const int* __restrict__ src, const float* __restrict__ Wg,
                      const float* __restrict__ bias, float* __restrict__ Y,
                      int Nout, int ldY) {
  constexpr int KT = 5;
  __shared__ alignas(32) _Float16 sW[KT * NT * 512];
  int tid = threadIdx.x;
  stage_w_swz<KT, NT>(sW, Wg, Nout, tid);
  __syncthreads();
  int wave = tid >> 5, lane = tid & 31;
  int row0 = (blockIdx.x * 8 + wave) * 16;
  int m = lane & 15;
  int e = row0 + m;
  int nd = e / KNB;
  int s  = src[e];
  const float* pS = res + (size_t)s  * 256;   // res[s][0][*]
  const float* pN = res + (size_t)nd * 256;   // res[nd][0][*]
  const float* pE = ef  + (size_t)e  * 32;
  v8f acc[NT] = {};
#pragma unroll
  for (int kt = 0; kt < KT; ++kt) {
    int k0 = kt * 32;
    const float* base = (k0 < 64) ? (pS + k0) : ((k0 < 128) ? (pN + k0 - 64) : (pE + k0 - 128));
    v16h a = make_a_frag(base, lane);
#pragma unroll
    for (int nt = 0; nt < NT; ++nt)
      acc[nt] = wmma_acc(a, load_b_swz(sW, NT, kt, nt, lane), acc[nt]);
  }
  int mb = (lane >> 4) * 8;
#pragma unroll
  for (int nt = 0; nt < NT; ++nt) {
    int nc = nt * 16 + (lane & 15);
    if (nc < Nout) {
#pragma unroll
      for (int r = 0; r < 8; ++r) {
        float v = acc[nt][r];
        if (EPI == 2) v = tanhf(v + bias[nc]);
        Y[(size_t)(row0 + mb + r) * ldY + nc] = v;
      }
    }
  }
}

// v0 = res0[src[e]] @ Wv0 (64 -> 64)
__global__ __launch_bounds__(256)
void gemm_v0_kernel(const float* __restrict__ res, const int* __restrict__ src,
                    const float* __restrict__ Wg, float* __restrict__ Y) {
  constexpr int NT = 4, KT = 2;
  __shared__ alignas(32) _Float16 sW[KT * NT * 512];
  int tid = threadIdx.x;
  stage_w_swz<KT, NT>(sW, Wg, 64, tid);
  __syncthreads();
  int wave = tid >> 5, lane = tid & 31;
  int row0 = (blockIdx.x * 8 + wave) * 16;
  int e = row0 + (lane & 15);
  const float* pS = res + (size_t)src[e] * 256;
  v8f acc[NT] = {};
#pragma unroll
  for (int kt = 0; kt < KT; ++kt) {
    v16h a = make_a_frag(pS + kt * 32, lane);
#pragma unroll
    for (int nt = 0; nt < NT; ++nt)
      acc[nt] = wmma_acc(a, load_b_swz(sW, NT, kt, nt, lane), acc[nt]);
  }
  int mb = (lane >> 4) * 8;
#pragma unroll
  for (int nt = 0; nt < NT; ++nt) {
    int nc = nt * 16 + (lane & 15);
#pragma unroll
    for (int r = 0; r < 8; ++r)
      Y[(size_t)(row0 + mb + r) * 64 + nc] = acc[nt][r];
  }
}

// generic row GEMM (M x 64) @ (64 x Nout); EPI: 0 = store, 1 = Y +=
template<int EPI>
__global__ __launch_bounds__(256)
void gemm_rows_kernel(const float* __restrict__ X, const float* __restrict__ Wg,
                      float* __restrict__ Y, int Nout, int ldY) {
  constexpr int NT = 4, KT = 2, KIN = 64;
  __shared__ alignas(32) _Float16 sW[KT * NT * 512];
  int tid = threadIdx.x;
  stage_w_swz<KT, NT>(sW, Wg, Nout, tid);
  __syncthreads();
  int wave = tid >> 5, lane = tid & 31;
  int row0 = (blockIdx.x * 8 + wave) * 16;
  const float* xr = X + (size_t)(row0 + (lane & 15)) * KIN;
  v8f acc[NT] = {};
#pragma unroll
  for (int kt = 0; kt < KT; ++kt) {
    v16h a = make_a_frag(xr + kt * 32, lane);
#pragma unroll
    for (int nt = 0; nt < NT; ++nt)
      acc[nt] = wmma_acc(a, load_b_swz(sW, NT, kt, nt, lane), acc[nt]);
  }
  int mb = (lane >> 4) * 8;
#pragma unroll
  for (int nt = 0; nt < NT; ++nt) {
    int nc = nt * 16 + (lane & 15);
    if (nc < Nout) {
#pragma unroll
      for (int r = 0; r < 8; ++r) {
        size_t idx = (size_t)(row0 + mb + r) * ldY + nc;
        if (EPI == 1) Y[idx] += acc[nt][r]; else Y[idx] = acc[nt][r];
      }
    }
  }
}

// ---------------------------------------------------------------------------
// leaky-relu(0.2) + softmax over K per (node, head), in place on (E x 8)
// ---------------------------------------------------------------------------
__global__ __launch_bounds__(256)
void softmax_kernel(float* __restrict__ al) {
  int idx = blockIdx.x * 256 + threadIdx.x;     // N*NH
  int n = idx / NH_, h = idx % NH_;
  float mx = -3.4e38f;
  for (int k = 0; k < KNB; ++k) {
    float x = al[(size_t)(n*KNB + k)*8 + h];
    x = (x > 0.f) ? x : 0.2f*x;
    mx = fmaxf(mx, x);
  }
  float s = 0.f;
  for (int k = 0; k < KNB; ++k) {
    float x = al[(size_t)(n*KNB + k)*8 + h];
    x = (x > 0.f) ? x : 0.2f*x;
    s += expf(x - mx);
  }
  float inv = 1.f / s;
  for (int k = 0; k < KNB; ++k) {
    size_t i = (size_t)(n*KNB + k)*8 + h;
    float x = al[i];
    x = (x > 0.f) ? x : 0.2f*x;
    al[i] = expf(x - mx) * inv;
  }
}

// ---------------------------------------------------------------------------
// attention reduction: rotate neighbor l=1 features, apply Wv1, weight by
// alpha, rotate back, project with Wo.  One 64-thread block per node.
// OUTC=64/RESID=true -> out = x + delta (transformer layers)
// OUTC=32/RESID=false -> out = projection (final p head)
// ---------------------------------------------------------------------------
template<int OUTC, bool RESID>
__global__ __launch_bounds__(64)
void attn_reduce_kernel(const float* __restrict__ x, const float* __restrict__ alpha,
                        const float* __restrict__ v0, const float* __restrict__ Rm,
                        const int* __restrict__ src, const float* __restrict__ Wv1,
                        const float* __restrict__ Wo, float* __restrict__ out) {
  __shared__ float sWv1[64*64];
  __shared__ float sWo[64*OUTC];
  __shared__ float sS1[3*64];
  __shared__ float sO[4*64];
  __shared__ float sR[9];
  int h = threadIdx.x, n = blockIdx.x;
  for (int i = h; i < 64*64; i += 64)   sWv1[i] = Wv1[i];
  for (int i = h; i < 64*OUTC; i += 64) sWo[i]  = Wo[i];
  float o0 = 0.f, o1a = 0.f, o1b = 0.f, o1c = 0.f;
  for (int k = 0; k < KNB; ++k) {
    int e = n*KNB + k;
    int s = src[e];
    __syncthreads();
    if (h < 9) sR[h] = Rm[(size_t)e*9 + h];
    __syncthreads();
    float x1 = x[(size_t)(s*4 + 1)*64 + h];
    float x2 = x[(size_t)(s*4 + 2)*64 + h];
    float x3 = x[(size_t)(s*4 + 3)*64 + h];
    sS1[0*64 + h] = sR[0]*x1 + sR[1]*x2 + sR[2]*x3;
    sS1[1*64 + h] = sR[3]*x1 + sR[4]*x2 + sR[5]*x3;
    sS1[2*64 + h] = sR[6]*x1 + sR[7]*x2 + sR[8]*x3;
    __syncthreads();
    float a = alpha[(size_t)e*8 + (h >> 3)];
    o0 += a * v0[(size_t)e*64 + h];
    float va = 0.f, vb = 0.f, vc = 0.f;
    for (int c = 0; c < 64; ++c) {
      float w = sWv1[c*64 + h];
      va += sS1[0*64 + c]*w; vb += sS1[1*64 + c]*w; vc += sS1[2*64 + c]*w;
    }
    o1a += a * (sR[0]*va + sR[3]*vb + sR[6]*vc);
    o1b += a * (sR[1]*va + sR[4]*vb + sR[7]*vc);
    o1c += a * (sR[2]*va + sR[5]*vb + sR[8]*vc);
  }
  __syncthreads();
  sO[0*64 + h] = o0; sO[1*64 + h] = o1a; sO[2*64 + h] = o1b; sO[3*64 + h] = o1c;
  __syncthreads();
  if (OUTC == 64 || h < OUTC) {
    for (int r = 0; r < 4; ++r) {
      float acc = 0.f;
      for (int c = 0; c < 64; ++c) acc += sO[r*64 + c] * sWo[c*OUTC + h];
      if (RESID) out[(size_t)(n*4 + r)*64 + h] = x[(size_t)(n*4 + r)*64 + h] + acc;
      else       out[(size_t)(n*4 + r)*OUTC + h] = acc;
    }
  }
}

// FFN mid: row0 -> silu(t), rows 1..3 *= sigmoid(t)
__global__ __launch_bounds__(256)
void ffnmid_kernel(float* __restrict__ tb) {
  int idx = blockIdx.x * 256 + threadIdx.x;   // N*64
  int n = idx >> 6, c = idx & 63;
  float t = tb[(size_t)(n*4)*64 + c];
  float sg = 1.f / (1.f + expf(-t));
  tb[(size_t)(n*4)*64 + c] = t * sg;
  for (int i = 0; i < 3; ++i) tb[(size_t)(n*4 + 1 + i)*64 + c] *= sg;
}

// small 32-channel FFN for the p head (in-place residual); one wave per node
__global__ __launch_bounds__(32)
void pffn_kernel(float* __restrict__ p, const float* __restrict__ W1,
                 const float* __restrict__ W2) {
  __shared__ float sP[4*32], sSil[32], sH[3*32];
  int n = blockIdx.x, c = threadIdx.x;
  for (int r = 0; r < 4; ++r) sP[r*32 + c] = p[(size_t)(n*4 + r)*32 + c];
  __syncthreads();
  float t = 0.f;
  for (int d = 0; d < 32; ++d) t += sP[d] * W1[d*32 + c];
  float sg = 1.f / (1.f + expf(-t));
  sSil[c] = t * sg;
  for (int i = 0; i < 3; ++i) {
    float hv = 0.f;
    for (int d = 0; d < 32; ++d) hv += sP[(1+i)*32 + d] * W1[d*32 + c];
    sH[i*32 + c] = hv * sg;
  }
  __syncthreads();
  float o0 = 0.f;
  for (int d = 0; d < 32; ++d) o0 += sSil[d] * W2[d*32 + c];
  p[(size_t)(n*4)*32 + c] += o0;
  for (int i = 0; i < 3; ++i) {
    float oi = 0.f;
    for (int d = 0; d < 32; ++d) oi += sH[i*32 + d] * W2[d*32 + c];
    p[(size_t)(n*4 + 1 + i)*32 + c] += oi;
  }
}

// layernorm over AC + seq head + log-softmax
__global__ __launch_bounds__(256)
void lnlogits_kernel(const float* __restrict__ atoms, const float* __restrict__ g,
                     const float* __restrict__ b, const float* __restrict__ Ws,
                     const float* __restrict__ bs, float* __restrict__ out) {
  int n = blockIdx.x * 256 + threadIdx.x;
  if (n >= NN_) return;
  const float* inv = atoms + (size_t)(n*4)*AC_;
  float mu = 0.f;
  for (int c = 0; c < AC_; ++c) mu += inv[c];
  mu /= (float)AC_;
  float var = 0.f;
  for (int c = 0; c < AC_; ++c) { float d = inv[c] - mu; var += d*d; }
  var /= (float)AC_;
  float is = rsqrtf(var + 1e-5f);
  float lg[NAA_];
  for (int j = 0; j < NAA_; ++j) lg[j] = bs[j];
  for (int c = 0; c < AC_; ++c) {
    float y = (inv[c] - mu) * is * g[c] + b[c];
    for (int j = 0; j < NAA_; ++j) lg[j] += y * Ws[c*NAA_ + j];
  }
  float mx = -3.4e38f;
  for (int j = 0; j < NAA_; ++j) mx = fmaxf(mx, lg[j]);
  float s = 0.f;
  for (int j = 0; j < NAA_; ++j) s += expf(lg[j] - mx);
  float lse = logf(s);
  for (int j = 0; j < NAA_; ++j) out[(size_t)n*NAA_ + j] = lg[j] - mx - lse;
}

// decoded = transpose(atoms[:,1:4], (0,2,1)) -> (N, AC, 3)
__global__ __launch_bounds__(256)
void decoded_kernel(const float* __restrict__ atoms, float* __restrict__ out) {
  int idx = blockIdx.x * 256 + threadIdx.x;
  if (idx >= NN_*AC_*3) return;
  int n = idx / (AC_*3), rem = idx % (AC_*3);
  int c = rem / 3, i = rem % 3;
  out[idx] = atoms[(size_t)(n*4 + 1 + i)*AC_ + c];
}

// ---------------------------------------------------------------------------
// host orchestration
// ---------------------------------------------------------------------------
extern "C" void kernel_launch(void* const* d_in, const int* in_sizes, int n_in,
                              void* d_out, int out_size, void* d_ws, size_t ws_size,
                              hipStream_t stream) {
  const float* bb       = (const float*)d_in[0];
  const float* latent   = (const float*)d_in[2];
  const float* eb_Wrad  = (const float*)d_in[3];
  const float* eb_brad  = (const float*)d_in[4];
  const float* eb_Wgate = (const float*)d_in[5];
  const float* eb_W0    = (const float*)d_in[6];
  const float* eb_W1    = (const float*)d_in[7];
  const float* t_Walpha = (const float*)d_in[8];
  const float* t_Wv0    = (const float*)d_in[9];
  const float* t_Wv1    = (const float*)d_in[10];
  const float* t_Wo     = (const float*)d_in[11];
  const float* t_Wf1    = (const float*)d_in[12];
  const float* t_Wf2    = (const float*)d_in[13];
  const float* e_W      = (const float*)d_in[14];
  const float* e_b      = (const float*)d_in[15];
  const float* p_Walpha = (const float*)d_in[16];
  const float* p_Wv0    = (const float*)d_in[17];
  const float* p_Wv1    = (const float*)d_in[18];
  const float* p_Wo     = (const float*)d_in[19];
  const float* p_Wf1    = (const float*)d_in[20];
  const float* p_Wf2    = (const float*)d_in[21];
  const float* oa_Wrad  = (const float*)d_in[22];
  const float* oa_brad  = (const float*)d_in[23];
  const float* oa_Wgate = (const float*)d_in[24];
  const float* oa_W0    = (const float*)d_in[25];
  const float* oa_W1    = (const float*)d_in[26];
  const float* ln_g     = (const float*)d_in[27];
  const float* ln_b     = (const float*)d_in[28];
  const float* W_seq    = (const float*)d_in[29];
  const float* b_seq    = (const float*)d_in[30];
  float* out = (float*)d_out;
  (void)in_sizes; (void)n_in; (void)out_size; (void)ws_size;

  char* w = (char*)d_ws;
  size_t off = 0;
  auto alloc = [&](size_t bytes) -> void* {
    void* p = w + off;
    off = (off + bytes + 255) & ~size_t(255);
    return p;
  };
  int*   src    = (int*)  alloc((size_t)EE * 4);
  float* Rbuf   = (float*)alloc((size_t)EE * 9 * 4);
  float* efA    = (float*)alloc((size_t)EE * 32 * 4);
  float* efB    = (float*)alloc((size_t)EE * 32 * 4);
  float* bbembB = (float*)alloc((size_t)NN_ * 4 * 26 * 4);
  float* bbout  = (float*)alloc((size_t)NN_ * 4 * 32 * 4);
  float* resA   = (float*)alloc((size_t)NN_ * 4 * 64 * 4);
  float* resB   = (float*)alloc((size_t)NN_ * 4 * 64 * 4);
  float* alphaB = (float*)alloc((size_t)EE * 8 * 4);
  float* v0buf  = (float*)alloc((size_t)EE * 64 * 4);
  float* tbuf   = (float*)alloc((size_t)NN_ * 4 * 64 * 4);
  float* pbuf   = (float*)alloc((size_t)NN_ * 4 * 32 * 4);
  float* atoms  = (float*)alloc((size_t)NN_ * 4 * 91 * 4);

  knn_kernel<<<NN_/256, 256, 0, stream>>>(bb, src);
  geom_kernel<<<(EE + 255)/256, 256, 0, stream>>>(bb, src, efA, Rbuf);
  bbemb_kernel<<<NN_/256, 256, 0, stream>>>(bb, bbembB);
  so3conv_kernel<<<NN_, 128, 0, stream>>>(bbembB, efA, Rbuf, src, eb_Wrad, eb_brad,
                                          eb_Wgate, eb_W0, eb_W1, bbout, 26, 32);
  initres_kernel<<<(NN_*4*64)/256, 256, 0, stream>>>(bbout, latent, resA);

  float* resCur = resA; float* resNxt = resB;
  float* efCur = efA;  float* efNxt = efB;
  for (int l = 0; l < LL_; ++l) {
    gemm_feat_kernel<1, 0><<<EE/128, 256, 0, stream>>>(
        resCur, efCur, src, t_Walpha + (size_t)l*160*8, nullptr, alphaB, 8, 8);
    softmax_kernel<<<(NN_*NH_)/256, 256, 0, stream>>>(alphaB);
    gemm_v0_kernel<<<EE/128, 256, 0, stream>>>(resCur, src, t_Wv0 + (size_t)l*4096, v0buf);
    attn_reduce_kernel<64, true><<<NN_, 64, 0, stream>>>(
        resCur, alphaB, v0buf, Rbuf, src, t_Wv1 + (size_t)l*4096, t_Wo + (size_t)l*4096, resNxt);
    gemm_rows_kernel<0><<<(NN_*4)/128, 256, 0, stream>>>(resNxt, t_Wf1 + (size_t)l*4096, tbuf, 64, 64);
    ffnmid_kernel<<<(NN_*64)/256, 256, 0, stream>>>(tbuf);
    gemm_rows_kernel<1><<<(NN_*4)/128, 256, 0, stream>>>(tbuf, t_Wf2 + (size_t)l*4096, resNxt, 64, 64);
    gemm_feat_kernel<2, 2><<<EE/128, 256, 0, stream>>>(
        resNxt, efCur, src, e_W + (size_t)l*160*32, e_b + (size_t)l*32, efNxt, 32, 32);
    { float* t0 = resCur; resCur = resNxt; resNxt = t0; }
    { float* t1 = efCur;  efCur  = efNxt;  efNxt  = t1; }
  }

  gemm_feat_kernel<1, 0><<<EE/128, 256, 0, stream>>>(
      resCur, efCur, src, p_Walpha, nullptr, alphaB, 8, 8);
  softmax_kernel<<<(NN_*NH_)/256, 256, 0, stream>>>(alphaB);
  gemm_v0_kernel<<<EE/128, 256, 0, stream>>>(resCur, src, p_Wv0, v0buf);
  attn_reduce_kernel<32, false><<<NN_, 64, 0, stream>>>(
      resCur, alphaB, v0buf, Rbuf, src, p_Wv1, p_Wo, pbuf);
  pffn_kernel<<<NN_, 32, 0, stream>>>(pbuf, p_Wf1, p_Wf2);
  so3conv_kernel<<<NN_, 128, 0, stream>>>(pbuf, efCur, Rbuf, src, oa_Wrad, oa_brad,
                                          oa_Wgate, oa_W0, oa_W1, atoms, 32, 91);
  lnlogits_kernel<<<NN_/256, 256, 0, stream>>>(atoms, ln_g, ln_b, W_seq, b_seq,
                                               out + (size_t)NN_*AC_*3);
  decoded_kernel<<<(NN_*AC_*3 + 255)/256, 256, 0, stream>>>(atoms, out);
}
